// MultiHeadAttentionBlock_12386685682425
// MI455X (gfx1250) — matrix operand order repaired and checked by reading, not compiled, not run on hardware
//
#include <hip/hip_runtime.h>

// ---------------------------------------------------------------------------
// MHA block for MI455X (gfx1250): f16 WMMA GEMMs + flash attention.
// B=2, S=2048, D=1024, H=16, DK=64.
// ---------------------------------------------------------------------------

typedef _Float16 h16;
typedef __attribute__((ext_vector_type(16))) _Float16 v16h;
typedef __attribute__((ext_vector_type(8)))  _Float16 v8h;
typedef __attribute__((ext_vector_type(8)))  float    v8f;

#define S_LEN   2048
#define DMODEL  1024
#define NHEAD   16
#define DKH     64
#define BATCH   2
#define MROWS   (BATCH * S_LEN)       // 4096
#define NEG_BIG (-3.0e38f)

__device__ __forceinline__ v8f wmma_f16(v16h a, v16h b, v8f c) {
  // D = A(16x32 f16) x B(32x16 f16) + C(16x16 f32)
  return __builtin_amdgcn_wmma_f32_16x16x32_f16(
      /*neg_a=*/false, a, /*neg_b=*/false, b,
      /*c_mod=*/(short)0, c, /*reuse_a=*/false, /*reuse_b=*/false);
}

__device__ __forceinline__ v16h cat8(v8h lo, v8h hi) {
  return __builtin_shufflevector(lo, hi, 0, 1, 2, 3, 4, 5, 6, 7,
                                         8, 9, 10, 11, 12, 13, 14, 15);
}

// ---------------------------------------------------------------------------
// f32 -> f16 conversion, 8 elements per thread (32B load / 16B store).
// ---------------------------------------------------------------------------
__global__ __launch_bounds__(256) void cvt_f32_f16(const float* __restrict__ src,
                                                   h16* __restrict__ dst, int n) {
  int i = (blockIdx.x * 256 + threadIdx.x) * 8;
  if (i + 8 > n) return;
  v8f a = *(const v8f*)(src + i);
  v8h o;
#pragma unroll
  for (int j = 0; j < 8; ++j) o[j] = (_Float16)a[j];
  *(v8h*)(dst + i) = o;
}

// ---------------------------------------------------------------------------
// GEMM: C[M,N] = A[M,K] * W^T, W stored [N,K] (torch Linear layout), K=N=1024.
// Block = 256 threads (8 waves), block tile 128x128, wave tile 32x64.
// MODE (compile-time): 0 = head-split f16 [b,h,s,dk]            (Q, K)
//                      1 = transposed head-split f16 [b,h,dk,s] (V)
//                      2 = plain f32 [m,n]                      (final output)
// ---------------------------------------------------------------------------
template <int MODE>
__global__ __launch_bounds__(256) void gemm_wmma(const h16* __restrict__ A,
                                                 const h16* __restrict__ W,
                                                 h16* __restrict__ outh,
                                                 float* __restrict__ outf) {
  const int tid  = threadIdx.x;
  const int wave = tid >> 5;
  const int lane = tid & 31;
  const int lm   = lane & 15;        // row/col within 16
  const int g    = lane >> 4;        // half-wave group
  const int lk8  = g * 8;            // A-frag K sub-offset

  const int mbase = blockIdx.y * 128 + (wave & 3) * 32;
  const int nbase = blockIdx.x * 128 + (wave >> 2) * 64;

  v8f acc[2][4];
#pragma unroll
  for (int i = 0; i < 2; ++i)
#pragma unroll
    for (int j = 0; j < 4; ++j) acc[i][j] = (v8f){0, 0, 0, 0, 0, 0, 0, 0};

  for (int kk = 0; kk < DMODEL; kk += 32) {
    v16h afr[2], bfr[4];
#pragma unroll
    for (int mi = 0; mi < 2; ++mi) {
      const h16* p = A + (size_t)(mbase + mi * 16 + lm) * DMODEL + kk;
      v8h lo = *(const v8h*)(p + 2 * lk8);
      v8h hi = *(const v8h*)(p + 2 * lk8 + 8);
      afr[mi] = cat8(lo, hi);
    }
#pragma unroll
    for (int ni = 0; ni < 4; ++ni) {
      const h16* p = W + (size_t)(nbase + ni * 16 + lm) * DMODEL + kk + g * 16;
      bfr[ni] = *(const v16h*)p;   // 16 contiguous K-halves: B-fragment
    }
    if (kk + 32 < DMODEL)
      __builtin_prefetch(A + (size_t)(mbase + lm) * DMODEL + kk + 32, 0, 1);
#pragma unroll
    for (int mi = 0; mi < 2; ++mi)
#pragma unroll
      for (int ni = 0; ni < 4; ++ni)
        acc[mi][ni] = wmma_f16(afr[mi], bfr[ni], acc[mi][ni]);
  }

  // Epilogue. D-layout: lane holds col n = nbase+ni*16+lm, row = base+r+8*g.
#pragma unroll
  for (int mi = 0; mi < 2; ++mi)
#pragma unroll
    for (int ni = 0; ni < 4; ++ni)
#pragma unroll
      for (int r = 0; r < 8; ++r) {
        int m = mbase + mi * 16 + r + 8 * g;
        int n = nbase + ni * 16 + lm;
        float v = acc[mi][ni][r];
        if (MODE == 0) {
          int b = m >> 11, s = m & (S_LEN - 1);
          int h = n >> 6,  dk = n & (DKH - 1);
          outh[(((size_t)(b * NHEAD + h) * S_LEN) + s) * DKH + dk] = (_Float16)v;
        } else if (MODE == 1) {
          int b = m >> 11, s = m & (S_LEN - 1);
          int h = n >> 6,  dk = n & (DKH - 1);
          outh[(((size_t)(b * NHEAD + h) * DKH) + dk) * S_LEN + s] = (_Float16)v;
        } else {
          outf[(size_t)m * DMODEL + n] = v;
        }
      }
}

// ---------------------------------------------------------------------------
// One 32-wide kv tile of flash attention against a 16-row q tile.
// MASK=true only for the peeled diagonal tile; coff = kvb - qbase there.
// ---------------------------------------------------------------------------
template <bool MASK>
__device__ __forceinline__ void attn_kv_tile(
    const h16* __restrict__ Kb, const h16* __restrict__ Vb,
    h16* __restrict__ pl, int kvb, int coff,
    const v16h (&aq)[2], v8f (&acc)[4],
    float (&rm)[8], float (&rl)[8],
    int lm, int g, int lk8) {
  // K^T B-fragments: lane holds kv row kvb+nj*16+lm, contiguous dk.
  v16h bk[2][2];
#pragma unroll
  for (int nj = 0; nj < 2; ++nj)
#pragma unroll
    for (int ks = 0; ks < 2; ++ks)
      bk[nj][ks] = *(const v16h*)(Kb + (size_t)(kvb + nj * 16 + lm) * DKH +
                                  ks * 32 + g * 16);

  v8f sc[2];
#pragma unroll
  for (int nj = 0; nj < 2; ++nj) {
    v8f s = (v8f){0, 0, 0, 0, 0, 0, 0, 0};
    s = wmma_f16(aq[0], bk[nj][0], s);
    s = wmma_f16(aq[1], bk[nj][1], s);
    sc[nj] = s;
  }

  // scale (+ causal mask on the diagonal tile)
#pragma unroll
  for (int nj = 0; nj < 2; ++nj)
#pragma unroll
    for (int r = 0; r < 8; ++r) {
      float v = sc[nj][r] * 0.125f;   // 1/sqrt(64)
      if (MASK) {
        int row = r + 8 * g;                // local q row, 0..15
        int col = nj * 16 + lm + coff;      // kv col relative to qbase
        if (col > row) v = NEG_BIG;
      }
      sc[nj][r] = v;
    }

  // online softmax update
  float tm[8], ps[8];
#pragma unroll
  for (int r = 0; r < 8; ++r)
    tm[r] = fmaxf(sc[0][r], sc[1][r]);
#pragma unroll
  for (int r = 0; r < 8; ++r)
    for (int off = 1; off < 16; off <<= 1)
      tm[r] = fmaxf(tm[r], __shfl_xor(tm[r], off));
#pragma unroll
  for (int r = 0; r < 8; ++r) {
    float nm    = fmaxf(rm[r], tm[r]);
    float alpha = __expf(rm[r] - nm);
    rm[r]  = nm;
    rl[r] *= alpha;
#pragma unroll
    for (int ni = 0; ni < 4; ++ni) acc[ni][r] *= alpha;
    float p0 = __expf(sc[0][r] - nm);
    float p1 = __expf(sc[1][r] - nm);
    sc[0][r] = p0;
    sc[1][r] = p1;
    ps[r] = p0 + p1;
  }
#pragma unroll
  for (int r = 0; r < 8; ++r) {
    for (int off = 1; off < 16; off <<= 1)
      ps[r] += __shfl_xor(ps[r], off);
    rl[r] += ps[r];
  }

  // P (D-layout regs) -> LDS -> A-layout fragment for P*V
#pragma unroll
  for (int nj = 0; nj < 2; ++nj)
#pragma unroll
    for (int r = 0; r < 8; ++r)
      pl[(r + 8 * g) * 32 + nj * 16 + lm] = (_Float16)sc[nj][r];
  asm volatile("s_wait_dscnt 0" ::: "memory");

  v16h ap;
  {
    const h16* p = pl + (size_t)lm * 32;
    ap = cat8(*(const v8h*)(p + 2 * lk8), *(const v8h*)(p + 2 * lk8 + 8));
  }

  // V^T B-fragments: lane holds dk col ni*16+lm, contiguous kv.
#pragma unroll
  for (int ni = 0; ni < 4; ++ni) {
    v16h bv = *(const v16h*)(Vb + (size_t)(ni * 16 + lm) * S_LEN + kvb + g * 16);
    acc[ni] = wmma_f16(ap, bv, acc[ni]);
  }
}

// ---------------------------------------------------------------------------
// Flash attention: one wave owns a 16-row query tile; kv loop in 32-col tiles.
// Q,K stored [b,h,s,dk] f16; V stored transposed [b,h,dk,s] f16.
// CTX written merged-head f16 [b,s,D] for the output projection.
// ---------------------------------------------------------------------------
__global__ __launch_bounds__(256) void attn_flash(const h16* __restrict__ Q,
                                                  const h16* __restrict__ Kh,
                                                  const h16* __restrict__ VT,
                                                  h16* __restrict__ CTX) {
  __shared__ h16 lds_p[8][16 * 32];   // per-wave 16x32 probability tile

  const int tid  = threadIdx.x;
  const int wave = tid >> 5;
  const int lane = tid & 31;
  const int lm   = lane & 15;
  const int g    = lane >> 4;
  const int lk8  = g * 8;

  const int bh = blockIdx.y;              // 0..31
  const int b  = bh >> 4, h = bh & 15;
  const int t  = blockIdx.x * 8 + wave;   // q-tile index, 0..127
  const int qbase = t * 16;

  const h16* Qb = Q  + (size_t)bh * S_LEN * DKH;
  const h16* Kb = Kh + (size_t)bh * S_LEN * DKH;
  const h16* Vb = VT + (size_t)bh * DKH * S_LEN;
  h16* pl = lds_p[wave];

  // Preload Q A-fragments: ks covers dk 0..31 / 32..63.
  v16h aq[2];
#pragma unroll
  for (int ks = 0; ks < 2; ++ks) {
    const h16* p = Qb + (size_t)(qbase + lm) * DKH + ks * 32;
    aq[ks] = cat8(*(const v8h*)(p + 2 * lk8), *(const v8h*)(p + 2 * lk8 + 8));
  }

  v8f acc[4];
#pragma unroll
  for (int j = 0; j < 4; ++j) acc[j] = (v8f){0, 0, 0, 0, 0, 0, 0, 0};

  float rm[8], rl[8];
#pragma unroll
  for (int r = 0; r < 8; ++r) { rm[r] = NEG_BIG; rl[r] = 0.0f; }

  // kv tiles: nfull fully-unmasked 32-col tiles, then one peeled masked tile
  // covering columns [nfull*32, qbase+16).
  const int nfull = (qbase + 15) / 32;    // == ntot - 1
  for (int kt = 0; kt < nfull; ++kt)
    attn_kv_tile<false>(Kb, Vb, pl, kt * 32, 0, aq, acc, rm, rl, lm, g, lk8);
  {
    const int kvb = nfull * 32;
    attn_kv_tile<true>(Kb, Vb, pl, kvb, kvb - qbase, aq, acc, rm, rl,
                       lm, g, lk8);
  }

  // normalize + write merged-head ctx
#pragma unroll
  for (int ni = 0; ni < 4; ++ni)
#pragma unroll
    for (int r = 0; r < 8; ++r) {
      int row = qbase + r + 8 * g;
      float v = acc[ni][r] / rl[r];
      CTX[((size_t)(b * S_LEN + row)) * DMODEL + h * DKH + ni * 16 + lm] =
          (_Float16)v;
    }
}

// ---------------------------------------------------------------------------
// Launch
// ---------------------------------------------------------------------------
extern "C" void kernel_launch(void* const* d_in, const int* in_sizes, int n_in,
                              void* d_out, int out_size, void* d_ws, size_t ws_size,
                              hipStream_t stream) {
  const float* q  = (const float*)d_in[0];
  const float* k  = (const float*)d_in[1];
  const float* v  = (const float*)d_in[2];
  const float* wq = (const float*)d_in[3];
  const float* wk = (const float*)d_in[4];
  const float* wv = (const float*)d_in[5];
  const float* wo = (const float*)d_in[6];
  // d_in[7] = mask, implied by causal logic; unused.
  float* out = (float*)d_out;

  const size_t DD = (size_t)DMODEL * DMODEL;   // 1M halves per weight
  const size_t SD = (size_t)MROWS * DMODEL;    // 4M halves per activation

  h16* ws    = (h16*)d_ws;
  h16* h_wq  = ws;
  h16* h_wk  = h_wq + DD;
  h16* h_wv  = h_wk + DD;
  h16* h_wo  = h_wv + DD;
  h16* h_xq  = h_wo + DD;
  h16* h_xk  = h_xq + SD;
  h16* h_xv  = h_xk + SD;
  h16* h_Q   = h_xv + SD;   // [b,h,s,dk]
  h16* h_K   = h_Q  + SD;   // [b,h,s,dk]
  h16* h_VT  = h_K  + SD;   // [b,h,dk,s]
  h16* h_CTX = h_VT + SD;   // [b,s,D]

  const int cw = (int)(DD / 8 / 256);   // 512 blocks per weight
  const int cx = (int)(SD / 8 / 256);   // 2048 blocks per activation
  cvt_f32_f16<<<cw, 256, 0, stream>>>(wq, h_wq, (int)DD);
  cvt_f32_f16<<<cw, 256, 0, stream>>>(wk, h_wk, (int)DD);
  cvt_f32_f16<<<cw, 256, 0, stream>>>(wv, h_wv, (int)DD);
  cvt_f32_f16<<<cw, 256, 0, stream>>>(wo, h_wo, (int)DD);
  cvt_f32_f16<<<cx, 256, 0, stream>>>(q, h_xq, (int)SD);
  cvt_f32_f16<<<cx, 256, 0, stream>>>(k, h_xk, (int)SD);
  cvt_f32_f16<<<cx, 256, 0, stream>>>(v, h_xv, (int)SD);

  dim3 ggrid(DMODEL / 128, MROWS / 128);   // (8, 32)
  gemm_wmma<0><<<ggrid, 256, 0, stream>>>(h_xq, h_wq, h_Q, nullptr);
  gemm_wmma<0><<<ggrid, 256, 0, stream>>>(h_xk, h_wk, h_K, nullptr);
  gemm_wmma<1><<<ggrid, 256, 0, stream>>>(h_xv, h_wv, h_VT, nullptr);

  dim3 agrid(S_LEN / 16 / 8, BATCH * NHEAD);   // (16, 32)
  attn_flash<<<agrid, 256, 0, stream>>>(h_Q, h_K, h_VT, h_CTX);

  gemm_wmma<2><<<ggrid, 256, 0, stream>>>(h_CTX, h_wo, nullptr, out);
}
